// Cross_Attention_Network_42339787604711
// MI455X (gfx1250) — matrix operand order repaired and checked by reading, not compiled, not run on hardware
//
#include <hip/hip_runtime.h>
#include <stdint.h>

typedef __bf16 bf16;
typedef __attribute__((ext_vector_type(16))) __bf16 v16bf;
typedef __attribute__((ext_vector_type(8)))  __bf16 v8bf;
typedef __attribute__((ext_vector_type(8)))  float  v8f;

#define N_B    8
#define C_DIM  512
#define L_DIM  2048
#define IC_DIM 256
#define M_DIM  (N_B * L_DIM)   /* 16384 */
#define BN_EPS 1e-5f

// ---------------------------------------------------------------------------
// BN statistics: per channel over (N, L); folds gamma/beta into scale+shift.
// grid = (C, 2)  y==0 -> A stats, y==1 -> B stats.  stats layout: [sA|tA|sB|tB]
// ---------------------------------------------------------------------------
__global__ __launch_bounds__(256) void bn_stats_kernel(
    const float* __restrict__ A, const float* __restrict__ B,
    const float* __restrict__ gamma, const float* __restrict__ beta,
    float* __restrict__ stats)
{
  const int c = blockIdx.x;
  const int which = blockIdx.y;
  const float* src = which ? B : A;
  const int t = threadIdx.x;
  float sum = 0.f, sq = 0.f;
  for (int n = 0; n < N_B; ++n) {
    const float* p = src + ((size_t)n * C_DIM + c) * L_DIM;
    #pragma unroll
    for (int i = 0; i < L_DIM / 256; ++i) {
      float x = p[t + i * 256];
      sum += x; sq += x * x;
    }
  }
  __shared__ float rs[256], rq[256];
  rs[t] = sum; rq[t] = sq;
  __syncthreads();
  for (int s = 128; s > 0; s >>= 1) {
    if (t < s) { rs[t] += rs[t + s]; rq[t] += rq[t + s]; }
    __syncthreads();
  }
  if (t == 0) {
    const float inv = 1.0f / (float)(N_B * L_DIM);
    float mean = rs[0] * inv;
    float var  = rq[0] * inv - mean * mean;
    float scl  = gamma[c] * rsqrtf(var + BN_EPS);
    float sft  = beta[c] - mean * scl;
    stats[which * 2 * C_DIM + c]         = scl;
    stats[which * 2 * C_DIM + C_DIM + c] = sft;
  }
}

// ---------------------------------------------------------------------------
// Fused BN*ReLU + transpose [N,C,L] -> [N,L,C], writing f32 (optional) + bf16.
// grid = (L/32, C/32, N), 32x32 LDS tile so both global accesses coalesce.
// ---------------------------------------------------------------------------
__global__ __launch_bounds__(256) void bn_apply_transpose_kernel(
    const float* __restrict__ X, const float* __restrict__ st,
    float* __restrict__ outF, bf16* __restrict__ outH)
{
  const int l0 = blockIdx.x * 32;
  const int c0 = blockIdx.y * 32;
  const int n  = blockIdx.z;
  __shared__ float tile[32][33];
  const int t = threadIdx.x;
  #pragma unroll
  for (int i = 0; i < 4; ++i) {
    int idx = t + i * 256;
    int ci = idx >> 5, li = idx & 31;
    float x = X[((size_t)n * C_DIM + c0 + ci) * L_DIM + l0 + li];
    float y = x * st[c0 + ci] + st[C_DIM + c0 + ci];
    tile[ci][li] = y > 0.f ? y : 0.f;
  }
  __syncthreads();
  #pragma unroll
  for (int i = 0; i < 4; ++i) {
    int idx = t + i * 256;
    int lo = idx >> 5, co = idx & 31;
    float y = tile[co][lo];
    size_t o = ((size_t)n * L_DIM + l0 + lo) * C_DIM + c0 + co;
    if (outF) outF[o] = y;
    outH[o] = (bf16)y;
  }
}

// ---------------------------------------------------------------------------
// Weight transpose + f32->bf16:  in [R,Cc] row-major  ->  out [Cc,R] row-major
// ---------------------------------------------------------------------------
__global__ __launch_bounds__(256) void wconv_kernel(
    const float* __restrict__ in, bf16* __restrict__ out, int R, int Cc)
{
  int idx = blockIdx.x * 256 + threadIdx.x;
  if (idx >= R * Cc) return;
  int r = idx / Cc, c = idx % Cc;
  out[(size_t)c * R + r] = (bf16)in[idx];
}

// ---------------------------------------------------------------------------
// Row softmax over 2048 columns, f32 in -> bf16 probabilities out.
// ---------------------------------------------------------------------------
__global__ __launch_bounds__(256) void softmax_kernel(
    const float* __restrict__ S, bf16* __restrict__ P)
{
  const int row = blockIdx.x, t = threadIdx.x;
  const float* sr = S + (size_t)row * L_DIM;
  float v[8];
  float mx = -3.402823466e38f;
  #pragma unroll
  for (int i = 0; i < 8; ++i) { v[i] = sr[t + i * 256]; mx = fmaxf(mx, v[i]); }
  __shared__ float red[256];
  red[t] = mx; __syncthreads();
  for (int s = 128; s > 0; s >>= 1) { if (t < s) red[t] = fmaxf(red[t], red[t + s]); __syncthreads(); }
  mx = red[0]; __syncthreads();
  float sum = 0.f;
  #pragma unroll
  for (int i = 0; i < 8; ++i) { v[i] = __expf(v[i] - mx); sum += v[i]; }
  red[t] = sum; __syncthreads();
  for (int s = 128; s > 0; s >>= 1) { if (t < s) red[t] += red[t + s]; __syncthreads(); }
  const float inv = 1.0f / red[0];
  bf16* pr = P + (size_t)row * L_DIM;
  #pragma unroll
  for (int i = 0; i < 8; ++i) pr[t + i * 256] = (bf16)(v[i] * inv);
}

// ---------------------------------------------------------------------------
// bf16 WMMA GEMM:  D[M,Nn] = A[M,K] * B[Nn,K]^T  (B given as [N][K] row-major)
// Block tile 128x128x32, 256 threads = 8 waves in 4(m) x 2(n); each wave owns
// a 32x64 tile = 2x4 v_wmma_f32_16x16x32_bf16 accumulators.
//
// Tiles are staged with the CDNA5 async global->LDS path:
//   global_load_async_to_lds_b128  (ASYNCcnt)  into double-buffered LDS,
//   s_wait_asynccnt 0 + workgroup barrier once per K-step, so stage s+1 loads
//   overlap stage s WMMA work.
//
// Fragment fetches use the same K order in A and B (not the ISA table order):
//   A and B: contiguous 16 K at 16*half.
//
// MODE 0: bf16 [M,N] (+optional bias)      MODE 1: bf16 transposed [b][n][l]
// MODE 2: f32 [M,N]                        MODE 4: f32 resid+bias, out [b][n][l]
// ---------------------------------------------------------------------------
#define BM 128
#define BN 128
#define BK 32
#define SA 40
#define SB 40

template<int MODE>
__global__ __launch_bounds__(256) void gemm_bf16_wmma(
    const bf16* __restrict__ A, const bf16* __restrict__ B,
    void* __restrict__ out, const float* __restrict__ bias,
    const float* __restrict__ resid, int M, int Nn, int K)
{
  __shared__ __align__(16) bf16 As[2][BM * SA];
  __shared__ __align__(16) bf16 Bs[2][BN * SB];
  const int tid  = threadIdx.x;
  const int wave = tid >> 5, lane = tid & 31;
  const int hlf  = lane >> 4, lm = lane & 15;
  const int wm   = (wave & 3) * 32;        // wave row offset within block tile
  const int wn   = (wave >> 2) * 64;       // wave col offset within block tile
  const int m0   = blockIdx.y * BM, n0 = blockIdx.x * BN;

  // global->LDS staging coordinates: 512 16B chunks per operand, 2 per thread
  const int rC = tid >> 2;                 // chunk row 0..63 (+64 for 2nd set)
  const int cC = (tid & 3) * 8;            // chunk col {0,8,16,24}

  // Issue one stage of async tile loads (4 async instructions per thread).
  auto issue_stage = [&](int s) {
    const int k0 = s * BK;
    bf16* abuf = As[s & 1];
    bf16* bbuf = Bs[s & 1];
    #pragma unroll
    for (int i = 0; i < 2; ++i) {
      int r = rC + i * 64;
      unsigned lds = (unsigned)(uintptr_t)&abuf[r * SA + cC];
      unsigned long long ga =
          (unsigned long long)(uintptr_t)&A[(size_t)(m0 + r) * K + k0 + cC];
      asm volatile("global_load_async_to_lds_b128 %0, %1, off"
                   :: "v"(lds), "v"(ga) : "memory");
    }
    #pragma unroll
    for (int i = 0; i < 2; ++i) {
      int r = rC + i * 64;
      unsigned lds = (unsigned)(uintptr_t)&bbuf[r * SB + cC];
      unsigned long long ga =
          (unsigned long long)(uintptr_t)&B[(size_t)(n0 + r) * K + k0 + cC];
      asm volatile("global_load_async_to_lds_b128 %0, %1, off"
                   :: "v"(lds), "v"(ga) : "memory");
    }
  };

  v8f acc[2][4] = {};
  const int S = K / BK;
  issue_stage(0);

  for (int s = 0; s < S; ++s) {
    // stage s data landed in LDS (own ops), then barrier so *all* waves' have.
    asm volatile("s_wait_asynccnt 0x0" ::: "memory");
    __syncthreads();
    if (s + 1 < S) issue_stage(s + 1);     // overlap next loads with compute

    const bf16* abuf = As[s & 1];
    const bf16* bbuf = Bs[s & 1];
    v16bf afr[2], bfr[4];
    #pragma unroll
    for (int i = 0; i < 2; ++i) {
      const bf16* base = &abuf[(wm + i * 16 + lm) * SA];
      union { v16bf v; v8bf h[2]; } u;
      u.h[0] = *(const v8bf*)(base + 16 * hlf);       // K = 16*hlf .. +8
      u.h[1] = *(const v8bf*)(base + 16 * hlf + 8);   // K = 16*hlf+8 .. +8
      afr[i] = u.v;
    }
    #pragma unroll
    for (int j = 0; j < 4; ++j) {
      const bf16* base = &bbuf[(wn + j * 16 + lm) * SB + 16 * hlf];
      union { v16bf v; v8bf h[2]; } u;
      u.h[0] = *(const v8bf*)(base);                  // K = 16*hlf .. +16
      u.h[1] = *(const v8bf*)(base + 8);
      bfr[j] = u.v;
    }
    #pragma unroll
    for (int i = 0; i < 2; ++i)
      #pragma unroll
      for (int j = 0; j < 4; ++j)
        acc[i][j] = __builtin_amdgcn_wmma_f32_16x16x32_bf16(
            false, afr[i], false, bfr[j], (short)0, acc[i][j], false, false);
    __syncthreads();   // all waves done reading buf (s&1) before it is refilled
  }

  // Epilogue. D layout: lane holds column n = lm; VGPR r is row wm+i*16+8*hlf+r.
  #pragma unroll
  for (int i = 0; i < 2; ++i) {
    #pragma unroll
    for (int j = 0; j < 4; ++j) {
      const int n = n0 + wn + j * 16 + lm;
      const float bval = bias ? bias[n] : 0.f;
      #pragma unroll
      for (int r = 0; r < 8; ++r) {
        const int m = m0 + wm + i * 16 + hlf * 8 + r;
        float val = acc[i][j][r] + bval;
        if (MODE == 0) {
          ((bf16*)out)[(size_t)m * Nn + n] = (bf16)val;
        } else if (MODE == 1) {           // v^T: [b][IC][L]
          ((bf16*)out)[((size_t)(m >> 11) * Nn + n) * L_DIM + (m & (L_DIM - 1))] = (bf16)val;
        } else if (MODE == 2) {
          ((float*)out)[(size_t)m * Nn + n] = val;
        } else {                          // MODE 4: residual + output transpose [b][C][L]
          float rv = resid[(size_t)m * Nn + n];
          ((float*)out)[((size_t)(m >> 11) * Nn + n) * L_DIM + (m & (L_DIM - 1))] = rv + val;
        }
      }
    }
  }
}

// ---------------------------------------------------------------------------
extern "C" void kernel_launch(void* const* d_in, const int* in_sizes, int n_in,
                              void* d_out, int out_size, void* d_ws, size_t ws_size,
                              hipStream_t stream)
{
  (void)in_sizes; (void)n_in; (void)out_size;
  const float* A       = (const float*)d_in[0];
  const float* B       = (const float*)d_in[1];
  const float* gamma   = (const float*)d_in[2];
  const float* beta    = (const float*)d_in[3];
  const float* theta_w = (const float*)d_in[4];
  const float* theta_b = (const float*)d_in[5];
  const float* phi_w   = (const float*)d_in[6];
  const float* phi_b   = (const float*)d_in[7];
  const float* g_w     = (const float*)d_in[8];
  const float* g_b     = (const float*)d_in[9];
  const float* W_w     = (const float*)d_in[10];
  const float* W_b     = (const float*)d_in[11];
  float* out = (float*)d_out;

  char* w = (char*)d_ws;
  size_t off = 0;
  auto take = [&](size_t bytes) -> char* {
    char* p = w + off;
    off = (off + bytes + 255) & ~(size_t)255;
    return p;
  };
  float* AnT = (float*)take((size_t)M_DIM * C_DIM * 4);   // BN(A) f32, [N,L,C]
  bf16*  AnH = (bf16*) take((size_t)M_DIM * C_DIM * 2);   // BN(A) bf16
  bf16*  BnH = (bf16*) take((size_t)M_DIM * C_DIM * 2);   // BN(B) bf16
  bf16*  Qb  = (bf16*) take((size_t)M_DIM * IC_DIM * 2);  // q
  bf16*  Kb  = (bf16*) take((size_t)M_DIM * IC_DIM * 2);  // k
  bf16*  VT  = (bf16*) take((size_t)M_DIM * IC_DIM * 2);  // v^T  [b][IC][L]
  bf16*  Yb  = (bf16*) take((size_t)M_DIM * IC_DIM * 2);  // attn @ v
  bf16*  ThT = (bf16*) take((size_t)IC_DIM * C_DIM * 2);  // theta_w^T
  bf16*  PhT = (bf16*) take((size_t)IC_DIM * C_DIM * 2);  // phi_w^T
  bf16*  GT  = (bf16*) take((size_t)IC_DIM * C_DIM * 2);  // g_w^T
  bf16*  WoT = (bf16*) take((size_t)C_DIM * IC_DIM * 2);  // W_w^T
  float* ST  = (float*)take((size_t)4 * C_DIM * 4);       // sA,tA,sB,tB
  float* SC  = (float*)take((size_t)L_DIM * L_DIM * 4);   // scores (reused per batch)
  bf16*  AT  = (bf16*) take((size_t)L_DIM * L_DIM * 2);   // softmax probs (reused)
  if (off > ws_size) return;

  // 1) BN statistics for A and B
  bn_stats_kernel<<<dim3(C_DIM, 2), 256, 0, stream>>>(A, B, gamma, beta, ST);

  // 2) BN*ReLU + transpose to [N,L,C]
  bn_apply_transpose_kernel<<<dim3(L_DIM / 32, C_DIM / 32, N_B), 256, 0, stream>>>(
      A, ST, AnT, AnH);
  bn_apply_transpose_kernel<<<dim3(L_DIM / 32, C_DIM / 32, N_B), 256, 0, stream>>>(
      B, ST + 2 * C_DIM, nullptr, BnH);

  // 3) weight transpose/convert (so GEMM B operand is [N][K] row-major bf16)
  const int wcnt = C_DIM * IC_DIM;
  wconv_kernel<<<(wcnt + 255) / 256, 256, 0, stream>>>(theta_w, ThT, C_DIM, IC_DIM);
  wconv_kernel<<<(wcnt + 255) / 256, 256, 0, stream>>>(phi_w,   PhT, C_DIM, IC_DIM);
  wconv_kernel<<<(wcnt + 255) / 256, 256, 0, stream>>>(g_w,     GT,  C_DIM, IC_DIM);
  wconv_kernel<<<(wcnt + 255) / 256, 256, 0, stream>>>(W_w,     WoT, IC_DIM, C_DIM);

  // 4) projections: q, k, v (v stored transposed for the attn@v GEMM)
  gemm_bf16_wmma<0><<<dim3(IC_DIM / BN, M_DIM / BM), 256, 0, stream>>>(
      AnH, ThT, Qb, theta_b, nullptr, M_DIM, IC_DIM, C_DIM);
  gemm_bf16_wmma<0><<<dim3(IC_DIM / BN, M_DIM / BM), 256, 0, stream>>>(
      BnH, PhT, Kb, phi_b, nullptr, M_DIM, IC_DIM, C_DIM);
  gemm_bf16_wmma<1><<<dim3(IC_DIM / BN, M_DIM / BM), 256, 0, stream>>>(
      AnH, GT, VT, g_b, nullptr, M_DIM, IC_DIM, C_DIM);

  // 5) per-batch attention (scores buffer reused; stream order serializes)
  for (int b = 0; b < N_B; ++b) {
    const bf16* qb = Qb + (size_t)b * L_DIM * IC_DIM;
    const bf16* kb = Kb + (size_t)b * L_DIM * IC_DIM;
    const bf16* vb = VT + (size_t)b * IC_DIM * L_DIM;
    bf16*       yb = Yb + (size_t)b * L_DIM * IC_DIM;
    gemm_bf16_wmma<2><<<dim3(L_DIM / BN, L_DIM / BM), 256, 0, stream>>>(
        qb, kb, SC, nullptr, nullptr, L_DIM, L_DIM, IC_DIM);
    softmax_kernel<<<L_DIM, 256, 0, stream>>>(SC, AT);
    gemm_bf16_wmma<0><<<dim3(IC_DIM / BN, L_DIM / BM), 256, 0, stream>>>(
        AT, vb, yb, nullptr, nullptr, L_DIM, IC_DIM, L_DIM);
  }

  // 6) output projection + residual + transpose back to [N,C,L]
  gemm_bf16_wmma<4><<<dim3(C_DIM / BN, M_DIM / BM), 256, 0, stream>>>(
      Yb, WoT, out, W_b, AnT, M_DIM, C_DIM, IC_DIM);
}
